// Graphormer2DAttentionBias_52347061404284
// MI455X (gfx1250) — compile-verified, matrix operand order checked by reading, not run on hardware
//
#include <hip/hip_runtime.h>
#include <hip/hip_bf16.h>
#include <stddef.h>

// ---------------------------------------------------------------------------
// Graphormer 2D attention bias for MI455X (gfx1250, wave32).
//
//  B=8, N=64, L=12, H=32 (LH=384), D=5, MAX_SPD=512, bond idx in [0,12).
//
//  The einsum row depends only on the bond-index triple (12^3 = 1728
//  possibilities), so we precompute T[d, triple, k] with a bf16 WMMA GEMM
//  (5 x [1728x384]x[384x384], f32 accum, 2.55 GFLOP = 19x fewer FLOPs than
//  the naive einsum) and the main pass becomes an L2-resident gather plus a
//  coalesced store of the 52 MB output (HBM floor ~2.3 us @ 23.3 TB/s).
// ---------------------------------------------------------------------------

typedef __attribute__((ext_vector_type(16))) __bf16 v16bf;
typedef __attribute__((ext_vector_type(8)))  float  v8f;

#define LH    384
#define NTRI  1728          // 12*12*12 distinct bond triples
#define NB    8
#define NN    64
#define ND    5
#define NP1   65

// ---------------------------------------------------------------------------
// Stage 1a: E[t, h] = bond0[i0,h] + bond1[i1,h] + bond2[i2,h]   (bf16)
// ---------------------------------------------------------------------------
__global__ void build_E(const float* __restrict__ b0, const float* __restrict__ b1,
                        const float* __restrict__ b2, __bf16* __restrict__ E) {
    int idx = blockIdx.x * blockDim.x + threadIdx.x;
    if (idx >= NTRI * LH) return;
    int t = idx / LH, h = idx % LH;
    int i0 = t / 144, i1 = (t / 12) % 12, i2 = t % 12;
    float v = b0[i0 * LH + h] + b1[i1 * LH + h] + b2[i2 * LH + h];
    E[idx] = (__bf16)v;
}

// ---------------------------------------------------------------------------
// Stage 1b: Wt[d, n, h] = edge_dis[d, h, n]  (bf16, N-major so B-fragments
// load as contiguous 16-byte chunks per lane)
// ---------------------------------------------------------------------------
__global__ void build_Wt(const float* __restrict__ W, __bf16* __restrict__ Wt) {
    int idx = blockIdx.x * blockDim.x + threadIdx.x;
    if (idx >= ND * LH * LH) return;
    int d = idx / (LH * LH);
    int r = idx % (LH * LH);
    int n = r / LH, h = r % LH;
    Wt[idx] = (__bf16)W[d * LH * LH + h * LH + n];
}

// ---------------------------------------------------------------------------
// Stage 2: T[d, t, k] = sum_h E[t,h] * edge_dis[d,h,k]  via WMMA bf16->f32.
//
// Register blocking: one wave owns a 32x64 output block (2 M-tiles x 4
// N-tiles, 8 x v8f accumulators = 64 VGPRs).  Per K-step: 4 A-chunk loads +
// 8 B-chunk loads (all b128) feed 8 back-to-back v_wmma_f32_16x16x32_bf16.
// 96 WMMAs per wave; operand traffic ~2.7x lower than 1-tile-per-wave.
//
// 16-bit operand fragment layout (wave32):
//   lanes 0-15 : row/col = lane,     K = kb+0..7  and kb+16..23
//   lanes 16-31: row/col = lane-16,  K = kb+8..15 and kb+24..31
// ---------------------------------------------------------------------------
#define MG (NTRI / 32)               // 54 M-groups (2 tiles each)
#define NG (LH / 64)                 // 6  N-groups (4 tiles each)
#define TOTAL_WAVES (ND * MG * NG)   // 1620

__global__ void gemm_T(const __bf16* __restrict__ E, const __bf16* __restrict__ Wt,
                       float* __restrict__ T) {
    int wave = blockIdx.x * (blockDim.x >> 5) + (threadIdx.x >> 5);
    if (wave >= TOTAL_WAVES) return;          // whole-wave uniform: EXEC stays all-1s
    int lane = threadIdx.x & 31;
    int d    = wave / (MG * NG);
    int rem  = wave % (MG * NG);
    int mg   = rem / NG;                      // m-tiles 2*mg, 2*mg+1
    int ng   = rem % NG;                      // n-tiles 4*ng .. 4*ng+3

    int half8 = (lane >> 4) * 8;              // 0 or 8
    const __bf16* Ea = E  + (size_t)(mg * 32 + (lane & 15)) * LH + half8;
    const __bf16* Wa = Wt + (size_t)d * LH * LH
                          + (size_t)(ng * 64 + (lane & 15)) * LH + half8;

    v8f acc[2][4];
    #pragma unroll
    for (int m = 0; m < 2; ++m)
        #pragma unroll
        for (int n = 0; n < 4; ++n)
            acc[m][n] = (v8f){};

    #pragma unroll
    for (int ks = 0; ks < 12; ++ks) {
        int kb = ks * 32;
        v16bf a[2], b[4];
        #pragma unroll
        for (int m = 0; m < 2; ++m) {
            const __bf16* p = Ea + (size_t)m * 16 * LH + kb;
            ((uint4*)&a[m])[0] = *(const uint4*)(p);
            ((uint4*)&a[m])[1] = *(const uint4*)(p + 16);
        }
        #pragma unroll
        for (int n = 0; n < 4; ++n) {
            const __bf16* p = Wa + (size_t)n * 16 * LH + kb;
            ((uint4*)&b[n])[0] = *(const uint4*)(p);
            ((uint4*)&b[n])[1] = *(const uint4*)(p + 16);
        }
        #pragma unroll
        for (int m = 0; m < 2; ++m)
            #pragma unroll
            for (int n = 0; n < 4; ++n)
                acc[m][n] = __builtin_amdgcn_wmma_f32_16x16x32_bf16(
                                false, a[m], false, b[n],
                                (short)0, acc[m][n], false, false);
    }

    // C/D layout: VGPR r -> M = r (lanes 0-15) or 8+r (lanes 16-31), N = lane&15
    int rbase = (lane < 16) ? 0 : 8;
    #pragma unroll
    for (int m = 0; m < 2; ++m) {
        float* Tp = T + (size_t)d * NTRI * LH + (size_t)(mg * 32 + m * 16) * LH;
        #pragma unroll
        for (int n = 0; n < 4; ++n) {
            int col = ng * 64 + n * 16 + (lane & 15);
            #pragma unroll
            for (int r = 0; r < 8; ++r)
                Tp[(size_t)(rbase + r) * LH + col] = acc[m][n][r];
        }
    }
}

// ---------------------------------------------------------------------------
// Stage 3: inner region out[b, lh, i+1, j+1].
// Block = (32 j, 32 lh) = 1024 threads so each gathered T-row cache line is
// fully consumed across 32 consecutive lh; stores are 32-float runs along j.
// Grid = (B*64*2 j-halves, 12 lh-groups).
// ---------------------------------------------------------------------------
__global__ void graphormer_inner(const float* __restrict__ attn_bias,
                                 const int*   __restrict__ spatial_pos,
                                 const int*   __restrict__ edge_input,
                                 const float* __restrict__ spd_emb,
                                 const float* __restrict__ T,
                                 float* __restrict__ out) {
    int bij = blockIdx.x;            // (b*64 + i) * 2 + jhalf
    int jh  = bij & 1;
    int bi  = bij >> 1;
    int b   = bi >> 6;
    int i   = bi & 63;
    int tx  = threadIdx.x;           // 0..31  (j within half)
    int ty  = threadIdx.y;           // 0..31  (lh within group)
    int j   = jh * 32 + tx;
    int lh  = blockIdx.y * 32 + ty;
    int tid = ty * 32 + tx;          // 0..1023

    __shared__ int   s_tri[32][ND];
    __shared__ float s_inv[32];
    __shared__ int   s_spd[32];

    if (tid < 32) {
        int jj  = jh * 32 + tid;
        int sp0 = spatial_pos[(b * NN + i) * NN + jj];
        int sp  = (sp0 == 0) ? 1 : sp0;
        if (sp > 1) sp -= 1;
        if (sp > ND) sp = ND;
        s_inv[tid] = 1.0f / (float)sp;
        s_spd[tid] = sp0;
    } else if (tid < 32 + 32 * ND) {
        int q  = tid - 32;
        int jj = q / ND, d = q % ND;
        const int* ep = edge_input
            + (size_t)((((b * NN + i) * NN + (jh * 32 + jj)) * ND + d)) * 3;
        int i0 = ep[0], i1 = ep[1], i2 = ep[2];
        i0 = (i0 > 11) ? 11 : i0;    // harness generates [0,12); clamp for safety
        i1 = (i1 > 11) ? 11 : i1;
        i2 = (i2 > 11) ? 11 : i2;
        s_tri[jj][d] = (i0 * 12 + i1) * 12 + i2;
    }
    __syncthreads();

    float esum = 0.0f;
    #pragma unroll
    for (int d = 0; d < ND; ++d)
        esum += T[((size_t)d * NTRI + s_tri[tx][d]) * LH + lh];

    float val  = spd_emb[(size_t)s_spd[tx] * LH + lh] + esum * s_inv[tx];
    float bias = attn_bias[(b * NP1 + (i + 1)) * NP1 + (j + 1)];
    out[(((size_t)b * LH + lh) * NP1 + (i + 1)) * NP1 + (j + 1)] = bias + val;
}

// ---------------------------------------------------------------------------
// Stage 4: borders — row 0 (all j) and column 0 (i>=1) get bias + graph_token.
// ---------------------------------------------------------------------------
__global__ void graphormer_border(const float* __restrict__ attn_bias,
                                  const float* __restrict__ graph_token,
                                  float* __restrict__ out) {
    int idx = blockIdx.x * blockDim.x + threadIdx.x;
    const int total = NB * LH * (NP1 + NN);    // 129 border cells per (b,lh)
    if (idx >= total) return;
    int r  = idx % (NP1 + NN);
    int lh = (idx / (NP1 + NN)) % LH;
    int b  = idx / ((NP1 + NN) * LH);
    int i, j;
    if (r < NP1) { i = 0; j = r; } else { i = r - NN; j = 0; }
    float v = attn_bias[(b * NP1 + i) * NP1 + j] + graph_token[lh];
    out[(((size_t)b * LH + lh) * NP1 + i) * NP1 + j] = v;
}

// ---------------------------------------------------------------------------
// Host launcher
// ---------------------------------------------------------------------------
extern "C" void kernel_launch(void* const* d_in, const int* in_sizes, int n_in,
                              void* d_out, int out_size, void* d_ws, size_t ws_size,
                              hipStream_t stream) {
    const float* attn_bias   = (const float*)d_in[0];
    const int*   spatial_pos = (const int*)  d_in[1];
    // d_in[2] = x  (unused by the reference output)
    const int*   edge_input  = (const int*)  d_in[3];
    const float* spd_emb     = (const float*)d_in[4];
    const float* graph_token = (const float*)d_in[5];
    const float* bond_emb0   = (const float*)d_in[6];
    const float* bond_emb1   = (const float*)d_in[7];
    const float* bond_emb2   = (const float*)d_in[8];
    const float* edge_dis    = (const float*)d_in[9];
    float* out = (float*)d_out;

    // Workspace layout (256B-aligned):
    //   E  : NTRI*LH bf16   = 1,327,104 B
    //   Wt : ND*LH*LH bf16  = 1,474,560 B
    //   T  : ND*NTRI*LH f32 = 13,271,040 B     (total ~16.1 MB, L2-resident)
    char* ws = (char*)d_ws;
    __bf16* E  = (__bf16*)(ws);
    __bf16* Wt = (__bf16*)(ws + 1327104);
    float*  T  = (float*) (ws + 1327104 + 1474560);

    // Stage 1: build bf16 operand tables
    {
        int n = NTRI * LH;
        build_E<<<(n + 255) / 256, 256, 0, stream>>>(bond_emb0, bond_emb1, bond_emb2, E);
    }
    {
        int n = ND * LH * LH;
        build_Wt<<<(n + 255) / 256, 256, 0, stream>>>(edge_dis, Wt);
    }

    // Stage 2: WMMA GEMM -> T   (1620 waves, 8 waves/block, 96 WMMAs/wave)
    {
        int blocks = (TOTAL_WAVES + 7) / 8;    // 203
        gemm_T<<<blocks, 256, 0, stream>>>(E, Wt, T);
    }

    // Stage 3: inner assembly (full cache-line use on T gathers,
    // coalesced 32-float stores along j)
    {
        dim3 grid(NB * NN * 2, LH / 32);       // (1024, 12)
        dim3 block(32, 32);                    // 1024 threads
        graphormer_inner<<<grid, block, 0, stream>>>(attn_bias, spatial_pos, edge_input,
                                                     spd_emb, T, out);
    }

    // Stage 4: borders
    {
        int n = NB * LH * (NP1 + NN);          // 396,288
        graphormer_border<<<(n + 255) / 256, 256, 0, stream>>>(attn_bias, graph_token, out);
    }
}
